// CrossDeformAttn_88184268522201
// MI455X (gfx1250) — matrix operand order repaired and checked by reading, not compiled, not run on hardware
//
#include <hip/hip_runtime.h>
#include <hip/hip_bf16.h>
#include <math.h>

// ---------------------------------------------------------------------------
// CDNA5 (gfx1250) implementation of CrossDeformAttn.
// bf16 WMMA (v_wmma_f32_16x16x32_bf16) for all GEMMs, f32 accumulate.
// ---------------------------------------------------------------------------

typedef __bf16 bf16;
typedef __attribute__((ext_vector_type(16))) __bf16 v16bf;
typedef __attribute__((ext_vector_type(8)))  __bf16 bf16x8;
typedef __attribute__((ext_vector_type(8)))  float  v8f;
typedef __attribute__((ext_vector_type(4)))  float  f32x4;

#define BB 8
#define LL 4096
#define DD 1024
#define HH 16
#define GG 4
#define DGc 256
#define DHc 64
#define KK 5

__device__ __forceinline__ v8f wmma_bf16(v16bf a, v16bf b, v8f c) {
  return __builtin_amdgcn_wmma_f32_16x16x32_bf16(
      /*neg_a=*/false, a, /*neg_b=*/false, b,
      /*c_mod=*/(short)0, c, /*reuse_a=*/false, /*reuse_b=*/false);
}

// A fragment (16 x 32 bf16). ISA layout: lane<16 -> row (lane&15),
// K = {0..7} U {16..23}; lane>=16 -> K = {8..15} U {24..31}.
__device__ __forceinline__ v16bf load_afrag(const bf16* row0, int lda, int lane, int k0) {
  const int r  = lane & 15;
  const int ko = (lane >> 4) << 3;               // 0 or 8
  const bf16* p = row0 + (size_t)r * lda + k0 + ko;
  union { v16bf v; bf16x8 h[2]; } u;
  u.h[0] = *(const bf16x8*)(p);
  u.h[1] = *(const bf16x8*)(p + 16);
  return u.v;
}

// B fragment (32 x 16 bf16), source stored row-major (N,K).
// ISA layout: lane<16 -> column (lane&15), K = {0..7} U {16..23}; lane>=16 -> K = {8..15} U {24..31}.
__device__ __forceinline__ v16bf load_bfrag(const bf16* col0, int ldb, int lane, int k0) {
  const int n  = lane & 15;
  const int ko = (lane >> 4) << 3;               // 0 or 8
  const bf16* p = col0 + (size_t)n * ldb + k0 + ko;
  union { v16bf v; bf16x8 h[2]; } u;
  u.h[0] = *(const bf16x8*)(p);
  u.h[1] = *(const bf16x8*)(p + 16);
  return u.v;
}

// ---------------------------------------------------------------------------
// cast f32 -> bf16, 8 elements per thread (b128 loads, b128 store)
// ---------------------------------------------------------------------------
__global__ void cast_f32_to_bf16_v8(const float* __restrict__ in,
                                    bf16* __restrict__ out, size_t n8) {
  size_t i = (size_t)blockIdx.x * blockDim.x + threadIdx.x;
  const size_t stride = (size_t)gridDim.x * blockDim.x;
  for (; i < n8; i += stride) {
    const f32x4 a = *(const f32x4*)(in + i * 8);
    const f32x4 b = *(const f32x4*)(in + i * 8 + 4);
    bf16x8 o;
#pragma unroll
    for (int e = 0; e < 4; ++e) { o[e] = (bf16)a[e]; o[e + 4] = (bf16)b[e]; }
    *(bf16x8*)(out + i * 8) = o;
  }
}

// ---------------------------------------------------------------------------
// Fold the two offset convs into a single (256 x 5) effective filter:
//   Weff[ci][k] = sum_co Woff2[co] * Woff1[co, ci, k]
//   beff        = boff2 + sum_co Woff2[co] * boff1[co]     (stored at [1280])
// ---------------------------------------------------------------------------
__global__ void build_weff(const float* __restrict__ Woff1,
                           const float* __restrict__ boff1,
                           const float* __restrict__ Woff2,
                           const float* __restrict__ boff2,
                           float* __restrict__ weff) {
  int t = blockIdx.x * blockDim.x + threadIdx.x;
  if (t < DGc * KK) {
    int ci = t / KK, k = t % KK;
    float s = 0.f;
    for (int co = 0; co < DGc; ++co)
      s += Woff2[co] * Woff1[(co * DGc + ci) * KK + k];
    weff[t] = s;
  }
  if (t == 0) {
    float s = boff2[0];
    for (int co = 0; co < DGc; ++co) s += Woff2[co] * boff1[co];
    weff[DGc * KK] = s;
  }
}

// ---------------------------------------------------------------------------
// Generic bf16 GEMM: C[m,n] = sum_k A[m,k]*B[n,k]  (+bias[m], +rel[m,n])
// A row-major (M,K), B row-major (N,K).  Block: 256 thr = 8 waves (2x4),
// each wave a 64x64 tile via 4x4 WMMA fragments.  Block tile 128x256.
// CM1: output M-dimension is unit-stride -> vectorized b128 stores.
// ---------------------------------------------------------------------------
template <typename OutT, bool HAS_REL, bool CM1>
__global__ __launch_bounds__(256)
void gemm_bf16_tn(const bf16* __restrict__ A, int lda, size_t strideAb,
                  const bf16* __restrict__ Bm, int ldb, size_t strideBb,
                  OutT* __restrict__ C, size_t strideCb, int strideCM, int strideCN,
                  const float* __restrict__ bias,
                  const float* __restrict__ rel, int relld, int Kdim) {
  const int lane = threadIdx.x & 31;
  const int wave = threadIdx.x >> 5;
  const int wm = wave >> 2, wn = wave & 3;
  const int m0 = blockIdx.y * 128 + wm * 64;
  const int n0 = blockIdx.x * 256 + wn * 64;
  const int b  = blockIdx.z;
  const bf16* Ab = A + (size_t)b * strideAb;
  const bf16* Bb = Bm + (size_t)b * strideBb;
  OutT* Cb = C + (size_t)b * strideCb;

  v8f acc[4][4] = {};
  for (int k0 = 0; k0 < Kdim; k0 += 32) {
    v16bf af[4], bfv[4];
#pragma unroll
    for (int i = 0; i < 4; ++i)
      af[i] = load_afrag(Ab + (size_t)(m0 + 16 * i) * lda, lda, lane, k0);
#pragma unroll
    for (int j = 0; j < 4; ++j)
      bfv[j] = load_bfrag(Bb + (size_t)(n0 + 16 * j) * ldb, ldb, lane, k0);
#pragma unroll
    for (int i = 0; i < 4; ++i)
#pragma unroll
      for (int j = 0; j < 4; ++j)
        acc[i][j] = wmma_bf16(af[i], bfv[j], acc[i][j]);
  }

  const int nl = lane & 15, hi = lane >> 4;
#pragma unroll
  for (int i = 0; i < 4; ++i)
#pragma unroll
    for (int j = 0; j < 4; ++j) {
      const int n = n0 + 16 * j + nl;
      const int mb = m0 + 16 * i + 8 * hi;     // 8 consecutive m values
      float vals[8];
#pragma unroll
      for (int r = 0; r < 8; ++r) {
        float v = acc[i][j][r] + bias[mb + r];
        if (HAS_REL) v += rel[(size_t)(mb + r) * relld + n];
        vals[r] = v;
      }
      if (CM1) {
        OutT* p = Cb + (size_t)n * strideCN + mb;
        if (sizeof(OutT) == 2) {               // one b128 store
          bf16x8 o;
#pragma unroll
          for (int r = 0; r < 8; ++r) o[r] = (bf16)vals[r];
          *(bf16x8*)p = o;
        } else {                               // two b128 stores
          f32x4 o0, o1;
#pragma unroll
          for (int r = 0; r < 4; ++r) { o0[r] = vals[r]; o1[r] = vals[r + 4]; }
          *(f32x4*)p = o0;
          *(f32x4*)(p + 4) = o1;
        }
      } else {
#pragma unroll
        for (int r = 0; r < 8; ++r)
          Cb[(size_t)(mb + r) * strideCM + (size_t)n * strideCN] = (OutT)vals[r];
      }
    }
}

// ---------------------------------------------------------------------------
// Offsets: off_core via folded filter, then tanh*K and grid_sample position.
// One thread per (bg, l).  q is bf16 (B, D, L).
// ---------------------------------------------------------------------------
__global__ __launch_bounds__(256)
void offsets_kernel(const bf16* __restrict__ qbf, const float* __restrict__ weff,
                    const float* __restrict__ boff2, float* __restrict__ pos) {
  __shared__ float wsh[DGc * KK + 1];
  for (int i = threadIdx.x; i < DGc * KK + 1; i += blockDim.x) wsh[i] = weff[i];
  __syncthreads();

  const int idx = blockIdx.x * blockDim.x + threadIdx.x;   // bg*L + l
  const int bg = idx >> 12;
  const int l  = idx & (LL - 1);
  const int b = bg >> 2, g = bg & 3;
  const bf16* qbase = qbf + ((size_t)b * DD + g * DGc) * LL;

  float raw;
  if (l < 2) {
    raw = boff2[0];                       // conv2 pad positions output bias
  } else {
    float s = wsh[DGc * KK];              // beff
    const int base = l - 4;               // (l-2) + k - 2
    for (int c = 0; c < DGc; ++c) {
      const bf16* qr = qbase + (size_t)c * LL;
      const float* wc = wsh + c * KK;
#pragma unroll
      for (int k = 0; k < KK; ++k) {
        const int lq = base + k;
        if (lq >= 0 && lq < LL) s += wc[k] * (float)qr[lq];
      }
    }
    raw = s;
  }
  const float off = tanhf(raw) * (float)KK;
  const float vs = 2.0f * ((float)l + off) / (float)(LL + 4 - 1) - 1.0f;
  pos[idx] = ((vs + 1.0f) * (float)LL - 1.0f) * 0.5f;
}

// ---------------------------------------------------------------------------
// Bilinear sample of x (f32, B,L,D) at pos -> xs (bf16, B,L,D).
// One thread handles 8 consecutive channels (same group g).
// ---------------------------------------------------------------------------
__global__ __launch_bounds__(256)
void sample_kernel(const float* __restrict__ x, const float* __restrict__ pos,
                   bf16* __restrict__ xs) {
  const size_t t = (size_t)blockIdx.x * blockDim.x + threadIdx.x;
  const int d8 = (int)(t & (DD / 8 - 1));
  const size_t bl = t >> 7;                // b*L + l
  const int l = (int)(bl & (LL - 1));
  const int b = (int)(bl >> 12);
  const int d = d8 * 8;
  const int g = d >> 8;

  const float p = pos[((size_t)(b * GG + g) << 12) + l];
  const float fi = floorf(p);
  const int i0 = (int)fi;
  const float w1 = p - fi, w0 = 1.0f - w1;
  const bool v0 = (i0 >= 0) && (i0 < LL);
  const bool v1 = (i0 + 1 >= 0) && (i0 + 1 < LL);
  const int c0 = min(max(i0, 0), LL - 1);
  const int c1 = min(max(i0 + 1, 0), LL - 1);
  const float* r0 = x + ((size_t)b * LL + c0) * DD + d;
  const float* r1 = x + ((size_t)b * LL + c1) * DD + d;

  bf16x8 o;
#pragma unroll
  for (int e = 0; e < 8; ++e) {
    float a = (v0 ? r0[e] : 0.f) * w0 + (v1 ? r1[e] : 0.f) * w1;
    o[e] = (bf16)a;
  }
  *(bf16x8*)(xs + bl * DD + d) = o;
}

// ---------------------------------------------------------------------------
// Channel attention per (b,h):  S = scale * qh khᵀ (contract over L=4096),
// softmax over j, out = attn vh.  q,k: bf16 (B,D,L); v,out: bf16 (B,L,D).
// Block = 128 threads (4 waves).  Partial S accumulated in LDS (ds_add_f32),
// softmax in LDS, second GEMM reads bf16 attn from LDS.
// ---------------------------------------------------------------------------
__global__ __launch_bounds__(128)
void attn_kernel(const bf16* __restrict__ q, const bf16* __restrict__ k,
                 const bf16* __restrict__ v, bf16* __restrict__ out) {
  __shared__ float S[64 * 64];
  __shared__ bf16  attnS[64 * 64];
  const int bh = blockIdx.x;
  const int b = bh >> 4, h = bh & 15;
  const int lane = threadIdx.x & 31;
  const int wave = threadIdx.x >> 5;      // 0..3
  const int nl = lane & 15, hiq = lane >> 4;

  for (int i = threadIdx.x; i < 64 * 64; i += 128) S[i] = 0.f;
  __syncthreads();

  const bf16* qb = q + ((size_t)b * DD + h * DHc) * LL;
  const bf16* kb = k + ((size_t)b * DD + h * DHc) * LL;

  {   // phase 1: partial S over this wave's K-slice of 1024
    v8f acc[4][4] = {};
    const int kbeg = wave * 1024, kend = kbeg + 1024;
    for (int k0 = kbeg; k0 < kend; k0 += 32) {
      v16bf af[4], bfv[4];
#pragma unroll
      for (int i = 0; i < 4; ++i)
        af[i] = load_afrag(qb + (size_t)(16 * i) * LL, LL, lane, k0);
#pragma unroll
      for (int j = 0; j < 4; ++j)
        bfv[j] = load_bfrag(kb + (size_t)(16 * j) * LL, LL, lane, k0);
#pragma unroll
      for (int i = 0; i < 4; ++i)
#pragma unroll
        for (int j = 0; j < 4; ++j)
          acc[i][j] = wmma_bf16(af[i], bfv[j], acc[i][j]);
    }
    const float scale = 0.03125f;         // D^-0.5 = 1/32
#pragma unroll
    for (int i = 0; i < 4; ++i)
#pragma unroll
      for (int j = 0; j < 4; ++j)
#pragma unroll
        for (int r = 0; r < 8; ++r) {
          const int mi = 16 * i + 8 * hiq + r;
          const int nj = 16 * j + nl;
          atomicAdd(&S[mi * 64 + nj], acc[i][j][r] * scale);
        }
  }
  __syncthreads();

  if (threadIdx.x < 64) {               // softmax over j for row i
    const int i = threadIdx.x;
    float mx = -INFINITY;
    for (int j = 0; j < 64; ++j) mx = fmaxf(mx, S[i * 64 + j]);
    float sum = 0.f;
    for (int j = 0; j < 64; ++j) sum += __expf(S[i * 64 + j] - mx);
    const float inv = 1.0f / sum;
    for (int j = 0; j < 64; ++j)
      attnS[i * 64 + j] = (bf16)(__expf(S[i * 64 + j] - mx) * inv);
  }
  __syncthreads();

  {   // phase 2: out[i, l] = sum_j attn[i,j] * v[l, j];  wave owns 1024 l's
    v16bf af2[2][4];
#pragma unroll
    for (int ks = 0; ks < 2; ++ks)
#pragma unroll
      for (int i = 0; i < 4; ++i)
        af2[ks][i] = load_afrag(attnS + (size_t)(16 * i) * 64, 64, lane, ks * 32);

    const bf16* vb = v + (size_t)b * LL * DD + h * DHc;
    bf16* ob = out + (size_t)b * LL * DD + h * DHc;

    for (int nt = 0; nt < 16; ++nt) {
      const int n0 = wave * 1024 + nt * 64;
      v8f acc[4][4] = {};
#pragma unroll
      for (int ks = 0; ks < 2; ++ks) {
        v16bf bfv[4];
#pragma unroll
        for (int j = 0; j < 4; ++j)
          bfv[j] = load_bfrag(vb + (size_t)(n0 + 16 * j) * DD, DD, lane, ks * 32);
#pragma unroll
        for (int i = 0; i < 4; ++i)
#pragma unroll
          for (int j = 0; j < 4; ++j)
            acc[i][j] = wmma_bf16(af2[ks][i], bfv[j], acc[i][j]);
      }
#pragma unroll
      for (int i = 0; i < 4; ++i)
#pragma unroll
        for (int j = 0; j < 4; ++j) {
          const int n = n0 + 16 * j + nl;
          const int mb = 16 * i + 8 * hiq;   // 8 consecutive m
          bf16x8 o;
#pragma unroll
          for (int r = 0; r < 8; ++r) o[r] = (bf16)acc[i][j][r];
          *(bf16x8*)(ob + (size_t)n * DD + mb) = o;   // one b128 store
        }
    }
  }
}

// ---------------------------------------------------------------------------
// launcher
// ---------------------------------------------------------------------------
extern "C" void kernel_launch(void* const* d_in, const int* in_sizes, int n_in,
                              void* d_out, int out_size, void* d_ws, size_t ws_size,
                              hipStream_t stream) {
  const float* x        = (const float*)d_in[0];
  const float* Wq       = (const float*)d_in[1];
  const float* bq       = (const float*)d_in[2];
  const float* Wk       = (const float*)d_in[3];
  const float* bk       = (const float*)d_in[4];
  const float* Wv       = (const float*)d_in[5];
  const float* bv       = (const float*)d_in[6];
  const float* Wo       = (const float*)d_in[7];
  const float* bo       = (const float*)d_in[8];
  const float* Woff1    = (const float*)d_in[9];
  const float* boff1    = (const float*)d_in[10];
  const float* Woff2    = (const float*)d_in[11];
  const float* boff2    = (const float*)d_in[12];
  const float* rel_bias = (const float*)d_in[13];
  float* outp = (float*)d_out;

  // ---- workspace carve-up ----
  char* w = (char*)d_ws;
  size_t off = 0;
  auto alloc = [&](size_t bytes) {
    void* p = w + off;
    off = (off + bytes + 255) & ~(size_t)255;
    return p;
  };
  const size_t actN = (size_t)BB * LL * DD;           // 33.5M elems
  bf16* x_bf   = (bf16*)alloc(actN * 2);              // (B,L,D)
  bf16* q_bf   = (bf16*)alloc(actN * 2);              // (B,D,L)
  bf16* xs_bf  = (bf16*)alloc(actN * 2);              // (B,L,D); reused as attn-out
  bf16* k_bf   = (bf16*)alloc(actN * 2);              // (B,D,L)
  bf16* v_bf   = (bf16*)alloc(actN * 2);              // (B,L,D)
  bf16* Wq_bf  = (bf16*)alloc((size_t)DD * DD * 2);
  bf16* Wk_bf  = (bf16*)alloc((size_t)DD * DD * 2);
  bf16* Wv_bf  = (bf16*)alloc((size_t)DD * DD * 2);
  bf16* Wo_bf  = (bf16*)alloc((size_t)DD * DD * 2);
  float* weff  = (float*)alloc((DGc * KK + 1) * 4);
  float* pos   = (float*)alloc((size_t)BB * GG * LL * 4);

  // ---- 1. casts (vectorized x8) ----
  cast_f32_to_bf16_v8<<<2048, 256, 0, stream>>>(x, x_bf, actN / 8);
  cast_f32_to_bf16_v8<<<512, 256, 0, stream>>>(Wq, Wq_bf, (size_t)DD * DD / 8);
  cast_f32_to_bf16_v8<<<512, 256, 0, stream>>>(Wk, Wk_bf, (size_t)DD * DD / 8);
  cast_f32_to_bf16_v8<<<512, 256, 0, stream>>>(Wv, Wv_bf, (size_t)DD * DD / 8);
  cast_f32_to_bf16_v8<<<512, 256, 0, stream>>>(Wo, Wo_bf, (size_t)DD * DD / 8);

  // ---- 2. fold offset convs ----
  build_weff<<<(DGc * KK + 255) / 256, 256, 0, stream>>>(Woff1, boff1, Woff2, boff2, weff);

  const dim3 gblk(256);
  const dim3 ggrd(LL / 256, DD / 128, BB);

  // ---- 3. Q = Wq x  -> q_bf (B,D,L) ----
  gemm_bf16_tn<bf16, false, false><<<ggrd, gblk, 0, stream>>>(
      Wq_bf, DD, 0, x_bf, DD, (size_t)LL * DD,
      q_bf, (size_t)DD * LL, /*CM*/ LL, /*CN*/ 1, bq, nullptr, 0, DD);

  // ---- 4. offsets -> sampling positions ----
  offsets_kernel<<<(BB * GG * LL) / 256, 256, 0, stream>>>(q_bf, weff, boff2, pos);

  // ---- 5. bilinear sample -> xs_bf (B,L,D) ----
  sample_kernel<<<(unsigned)(actN / 8 / 256), 256, 0, stream>>>(x, pos, xs_bf);

  // ---- 6. K = Wk xs -> k_bf (B,D,L) ----
  gemm_bf16_tn<bf16, false, false><<<ggrd, gblk, 0, stream>>>(
      Wk_bf, DD, 0, xs_bf, DD, (size_t)LL * DD,
      k_bf, (size_t)DD * LL, /*CM*/ LL, /*CN*/ 1, bk, nullptr, 0, DD);

  // ---- 7. V = Wv xs + rel_bias -> v_bf (B,L,D) ----
  gemm_bf16_tn<bf16, true, true><<<ggrd, gblk, 0, stream>>>(
      Wv_bf, DD, 0, xs_bf, DD, (size_t)LL * DD,
      v_bf, (size_t)LL * DD, /*CM*/ 1, /*CN*/ DD, bv, rel_bias, LL, DD);

  // ---- 8. channel attention -> attn-out in xs_bf (B,L,D) ----
  attn_kernel<<<BB * HH, 128, 0, stream>>>(q_bf, k_bf, v_bf, xs_bf);

  // ---- 9. final projection -> d_out (B,L,D) f32 ----
  gemm_bf16_tn<float, false, true><<<ggrd, gblk, 0, stream>>>(
      Wo_bf, DD, 0, xs_bf, DD, (size_t)LL * DD,
      outp, (size_t)LL * DD, /*CM*/ 1, /*CN*/ DD, bo, nullptr, 0, DD);

  (void)in_sizes; (void)n_in; (void)out_size; (void)ws_size;
}
